// LlamaPrefixDecoderLayer_3478923509896
// MI455X (gfx1250) — matrix operand order repaired and checked by reading, not compiled, not run on hardware
//
#include <hip/hip_runtime.h>
#include <hip/hip_bf16.h>

// ---------------- problem constants ----------------
constexpr int Bc  = 2;
constexpr int Qc  = 2048;
constexpr int Pc  = 128;
constexpr int Hc  = 2048;
constexpr int NHc = 16;
constexpr int HDc = 128;
constexpr int Ic  = 5632;
constexpr int Sc  = Qc + Pc;          // 2176 combined sequence
constexpr float EPSc = 1e-5f;

typedef __bf16 v16bf __attribute__((ext_vector_type(16)));
typedef float  v8f   __attribute__((ext_vector_type(8)));
typedef int    v4i_vs __attribute__((vector_size(16)));   // builtin's pointee type

union Frag { v16bf v; uint4 q[2]; };

// ---------------- CDNA5 async global->LDS helpers ----------------
// Only enabled in the gfx1250 device pass; host pass / missing builtin falls
// back to the synchronous staging path.
#if defined(__gfx1250__) && __has_builtin(__builtin_amdgcn_global_load_async_to_lds_b128)
#define HAVE_ASYNC_LDS 1
#else
#define HAVE_ASYNC_LDS 0
#endif

__device__ __forceinline__ void async_copy16(const void* g, void* l) {
#if HAVE_ASYNC_LDS
    __builtin_amdgcn_global_load_async_to_lds_b128(
        (__attribute__((address_space(1))) v4i_vs*)g,
        (__attribute__((address_space(3))) v4i_vs*)l, 0, 0);
#else
    (void)g; (void)l;
#endif
}

__device__ __forceinline__ void wait_async_zero() {
#if HAVE_ASYNC_LDS
#if __has_builtin(__builtin_amdgcn_s_wait_asynccnt)
    __builtin_amdgcn_s_wait_asynccnt(0);
#else
    asm volatile("s_wait_asynccnt 0x0" ::: "memory");
#endif
#endif
}

// ---------------- f32 -> bf16 cast (weights) ----------------
__global__ void cast_f32_bf16_kernel(const float* __restrict__ in,
                                     __bf16* __restrict__ out, size_t n) {
    size_t i = (size_t)blockIdx.x * blockDim.x + threadIdx.x;
    size_t stride = (size_t)gridDim.x * blockDim.x;
    for (; i < n; i += stride) out[i] = (__bf16)in[i];
}

// ---------------- RMSNorm (+ cast to bf16) ----------------
__global__ __launch_bounds__(256)
void rmsnorm_cast_kernel(const float* __restrict__ srcA,
                         const float* __restrict__ srcB,
                         const float* __restrict__ w,
                         __bf16* __restrict__ out, int split) {
    int row = blockIdx.x;
    const float* src;
    if (split) {
        int b = row / Sc, s = row % Sc;
        src = (s < Qc) ? srcA + (size_t)(b * Qc + s) * Hc
                       : srcB + (size_t)(b * Pc + (s - Qc)) * Hc;
    } else {
        src = srcA + (size_t)row * Hc;
    }
    int tid = threadIdx.x;
    float ss = 0.f;
    for (int i = tid; i < Hc; i += 256) { float v = src[i]; ss += v * v; }
    for (int off = 16; off; off >>= 1) ss += __shfl_xor(ss, off, 32);
    __shared__ float red[8];
    if ((tid & 31) == 0) red[tid >> 5] = ss;
    __syncthreads();
    float tot = 0.f;
#pragma unroll
    for (int i = 0; i < 8; ++i) tot += red[i];
    float inv = rsqrtf(tot / (float)Hc + EPSc);
    for (int i = tid; i < Hc; i += 256)
        out[(size_t)row * Hc + i] = (__bf16)(src[i] * inv * w[i]);
}

// ---------------- WMMA GEMM: C[M,N] = A[M,K] * W[N,K]^T ----------------
// EPI 0: store bf16. EPI 1: f32 = acc + split-residual. EPI 2: final split out.
// Async path: double-buffered LDS, GLOBAL_LOAD_ASYNC_TO_LDS_B128 prefetch of
// tile kt+1 overlapped with WMMA on tile kt.
template<int EPI>
__global__ __launch_bounds__(256)
void gemm_bf16_wmma_kernel(const __bf16* __restrict__ A,
                           const __bf16* __restrict__ W,
                           int M, int N, int K,
                           __bf16* __restrict__ Obf,
                           float* __restrict__ Of,
                           const float* __restrict__ resA,
                           const float* __restrict__ resB,
                           const float* __restrict__ addf,
                           float* __restrict__ outSplit) {
#if HAVE_ASYNC_LDS
    __shared__ __align__(16) __bf16 AsBuf[2][128 * 32];
    __shared__ __align__(16) __bf16 BsBuf[2][128 * 32];
#else
    __shared__ __align__(16) __bf16 AsBuf[1][128 * 32];
    __shared__ __align__(16) __bf16 BsBuf[1][128 * 32];
#endif

    const int tid  = threadIdx.x;
    const int wave = tid >> 5, lane = tid & 31;
    const int half = lane >> 4, l16 = lane & 15;
    const int wm = (wave >> 1) * 32;   // wave M offset in tile
    const int wn = (wave & 1) * 64;    // wave N offset in tile
    const size_t m0 = (size_t)blockIdx.y * 128;
    const size_t n0 = (size_t)blockIdx.x * 128;

    v8f zero = {};
    v8f acc[2][4];
#pragma unroll
    for (int mi = 0; mi < 2; ++mi)
#pragma unroll
        for (int ni = 0; ni < 4; ++ni) acc[mi][ni] = zero;

    const int KT = K / 32;
    const int srow = tid >> 2, scol = tid & 3;      // 64 rows x 4 chunks per pass
#if HAVE_ASYNC_LDS
    // prologue: async-stage tile 0 into buffer 0
#pragma unroll
    for (int p = 0; p < 2; ++p) {
        int row = srow + p * 64;
        async_copy16(A + (m0 + row) * (size_t)K + scol * 8, AsBuf[0] + row * 32 + scol * 8);
        async_copy16(W + (n0 + row) * (size_t)K + scol * 8, BsBuf[0] + row * 32 + scol * 8);
    }
    wait_async_zero();
    __syncthreads();
#endif

    for (int kt = 0; kt < KT; ++kt) {
#if HAVE_ASYNC_LDS
        const int cur = kt & 1;
        const __bf16* as = AsBuf[cur];
        const __bf16* bs = BsBuf[cur];
        if (kt + 1 < KT) {   // async prefetch next tile into other buffer
            __bf16* an = AsBuf[cur ^ 1];
            __bf16* bn = BsBuf[cur ^ 1];
            size_t ko = (size_t)(kt + 1) * 32;
#pragma unroll
            for (int p = 0; p < 2; ++p) {
                int row = srow + p * 64;
                async_copy16(A + (m0 + row) * (size_t)K + ko + scol * 8, an + row * 32 + scol * 8);
                async_copy16(W + (n0 + row) * (size_t)K + ko + scol * 8, bn + row * 32 + scol * 8);
            }
        }
#else
        const __bf16* as = AsBuf[0];
        const __bf16* bs = BsBuf[0];
#pragma unroll
        for (int p = 0; p < 2; ++p) {
            int row = srow + p * 64;
            const uint4* sa = reinterpret_cast<const uint4*>(
                A + (m0 + row) * (size_t)K + (size_t)kt * 32) + scol;
            reinterpret_cast<uint4*>(AsBuf[0] + row * 32)[scol] = *sa;
            const uint4* sb = reinterpret_cast<const uint4*>(
                W + (n0 + row) * (size_t)K + (size_t)kt * 32) + scol;
            reinterpret_cast<uint4*>(BsBuf[0] + row * 32)[scol] = *sb;
        }
        __syncthreads();
        if (kt + 1 < KT && tid < 128) {
            __builtin_prefetch(A + (m0 + tid) * (size_t)K + (size_t)(kt + 1) * 32, 0, 1);
            __builtin_prefetch(W + (n0 + tid) * (size_t)K + (size_t)(kt + 1) * 32, 0, 1);
        }
#endif

        Frag a[2], b[4];
#pragma unroll
        for (int mi = 0; mi < 2; ++mi) {
            const uint4* ap = reinterpret_cast<const uint4*>(as + (wm + mi * 16 + l16) * 32);
            a[mi].q[0] = ap[half];      // K = half*8 .. +8
            a[mi].q[1] = ap[half + 2];  // K = 16 + half*8 .. +8
        }
#pragma unroll
        for (int ni = 0; ni < 4; ++ni) {
            const uint4* bp = reinterpret_cast<const uint4*>(
                bs + (wn + ni * 16 + l16) * 32);
            b[ni].q[0] = bp[half];
            b[ni].q[1] = bp[half + 2];
        }
#pragma unroll
        for (int mi = 0; mi < 2; ++mi)
#pragma unroll
            for (int ni = 0; ni < 4; ++ni)
                acc[mi][ni] = __builtin_amdgcn_wmma_f32_16x16x32_bf16(
                    false, a[mi].v, false, b[ni].v, (short)0, acc[mi][ni], false, false);

#if HAVE_ASYNC_LDS
        wait_async_zero();   // prefetch done before next iteration reads it
        __syncthreads();
#else
        __syncthreads();
#endif
    }

    // epilogue: C layout -> row = r + 8*half, col = lane%16 within 16x16 tile
#pragma unroll
    for (int mi = 0; mi < 2; ++mi) {
#pragma unroll
        for (int ni = 0; ni < 4; ++ni) {
#pragma unroll
            for (int r = 0; r < 8; ++r) {
                size_t row = m0 + wm + mi * 16 + r + 8 * half;
                size_t col = n0 + wn + ni * 16 + l16;
                float v = acc[mi][ni][r];
                if (EPI == 0) {
                    Obf[row * (size_t)N + col] = (__bf16)v;
                } else if (EPI == 1) {
                    int b = (int)(row / Sc), s = (int)(row % Sc);
                    float res = (s < Qc)
                        ? resA[(size_t)(b * Qc + s) * Hc + col]
                        : resB[(size_t)(b * Pc + (s - Qc)) * Hc + col];
                    Of[row * (size_t)N + col] = res + v;
                } else {
                    int b = (int)(row / Sc), s = (int)(row % Sc);
                    float o = addf[row * (size_t)N + col] + v;
                    if (s < Qc)
                        outSplit[(size_t)(b * Qc + s) * Hc + col] = o;
                    else
                        outSplit[(size_t)(Bc * Qc) * Hc +
                                 (size_t)(b * Pc + (s - Qc)) * Hc + col] = o;
                }
            }
        }
    }
}

// ---------------- RoPE (in-place on bf16 [B*S, H]) ----------------
__global__ void rope_kernel(__bf16* __restrict__ buf,
                            const int* __restrict__ pos_m,
                            const int* __restrict__ pos_p) {
    size_t idx = (size_t)blockIdx.x * blockDim.x + threadIdx.x;
    size_t total = (size_t)Bc * Sc * NHc * (HDc / 2);
    if (idx >= total) return;
    int i = (int)(idx & 63);
    size_t t = idx >> 6;
    int h = (int)(t % NHc);
    size_t tok = t / NHc;
    int s = (int)(tok % Sc);
    int pos = (s < Qc) ? pos_m[s] : pos_p[s - Qc];
    float freq = __powf(10000.0f, -(2.0f * i) / (float)HDc);
    float ang = (float)pos * freq;
    float sn, cs;
    __sincosf(ang, &sn, &cs);
    size_t base = tok * Hc + (size_t)h * HDc;
    float x1 = (float)buf[base + i];
    float x2 = (float)buf[base + i + 64];
    buf[base + i]      = (__bf16)(x1 * cs - x2 * sn);
    buf[base + i + 64] = (__bf16)(x2 * cs + x1 * sn);
}

// ---------------- Flash attention with WMMA ----------------
// grid: (S/64, NH, B), block: 128 (4 waves, 16 queries each)
__global__ __launch_bounds__(128)
void attn_wmma_kernel(const __bf16* __restrict__ qb,
                      const __bf16* __restrict__ kb,
                      const __bf16* __restrict__ vb,
                      const float* __restrict__ mm,
                      const float* __restrict__ mp,
                      __bf16* __restrict__ ob) {
    __shared__ __align__(16) __bf16 Ks[32 * 128];      // 32 keys x HD
    __shared__ __align__(16) __bf16 Vt[128 * 32];      // HD x 32 keys (transposed)
    __shared__ __align__(16) __bf16 Ps[4][16 * 32];    // per-wave probs 16x32

    const int tid  = threadIdx.x;
    const int wave = tid >> 5, lane = tid & 31;
    const int half = lane >> 4, l16 = lane & 15;
    const int b = blockIdx.z, h = blockIdx.y;
    const int q0 = blockIdx.x * 64 + wave * 16;
    const float scale = 0.08838834764831845f;  // 1/sqrt(128)

    // preload Q fragments for all 4 HD k-steps
    Frag aq[4];
    {
        const __bf16* qrow = qb + ((size_t)(b * Sc + q0 + l16) * Hc + (size_t)h * HDc);
#pragma unroll
        for (int ks = 0; ks < 4; ++ks) {
            const uint4* p = reinterpret_cast<const uint4*>(qrow + ks * 32);
            aq[ks].q[0] = p[half];
            aq[ks].q[1] = p[half + 2];
        }
    }

    v8f zero = {};
    v8f acc[8];
#pragma unroll
    for (int nc = 0; nc < 8; ++nc) acc[nc] = zero;
    float lrow[8];
#pragma unroll
    for (int r = 0; r < 8; ++r) lrow[r] = 0.f;

    const int qg = q0 + 8 * half;   // C-layout row base: row = qg + r

    for (int kt = 0; kt < Sc / 32; ++kt) {
        const int kbase = kt * 32;
        // stage K rows (async direct-to-LDS) and V transposed (register path)
#pragma unroll
        for (int p = 0; p < 4; ++p) {
            int e = tid + p * 128;
            int row = e >> 4, c = e & 15;
            size_t gsrc = (size_t)(b * Sc + kbase + row) * Hc + (size_t)h * HDc;
#if HAVE_ASYNC_LDS
            async_copy16(kb + gsrc + (size_t)c * 8, Ks + row * 128 + c * 8);
#else
            const uint4* ksrc = reinterpret_cast<const uint4*>(kb + gsrc) + c;
            reinterpret_cast<uint4*>(Ks + row * 128)[c] = *ksrc;
#endif
            uint4 vv = *(reinterpret_cast<const uint4*>(vb + gsrc) + c);
            const __bf16* ve = reinterpret_cast<const __bf16*>(&vv);
#pragma unroll
            for (int j = 0; j < 8; ++j) Vt[(c * 8 + j) * 32 + row] = ve[j];
        }
        wait_async_zero();
        __syncthreads();

        // QK^T: two 16-key score tiles, contraction over HD=128 (4 wmma each)
#pragma unroll
        for (int nk = 0; nk < 2; ++nk) {
            v8f sc = zero;
#pragma unroll
            for (int ks = 0; ks < 4; ++ks) {
                Frag fb;
                const uint4* bp = reinterpret_cast<const uint4*>(
                    Ks + (nk * 16 + l16) * 128 + ks * 32);
                fb.q[0] = bp[half];
                fb.q[1] = bp[half + 2];
                sc = __builtin_amdgcn_wmma_f32_16x16x32_bf16(
                    false, aq[ks].v, false, fb.v, (short)0, sc, false, false);
            }
            int key = kbase + nk * 16 + l16;
#pragma unroll
            for (int r = 0; r < 8; ++r) {
                int qrow = qg + r;
                float mval = (qrow < Qc)
                    ? mm[(size_t)(b * Qc + qrow) * Sc + key]
                    : mp[(size_t)(b * Pc + (qrow - Qc)) * Sc + key];
                float pv = __expf(sc[r] * scale + mval);
                float rs = pv;
                rs += __shfl_xor(rs, 1, 32);
                rs += __shfl_xor(rs, 2, 32);
                rs += __shfl_xor(rs, 4, 32);
                rs += __shfl_xor(rs, 8, 32);
                lrow[r] += rs;
                Ps[wave][(r + 8 * half) * 32 + nk * 16 + l16] = (__bf16)pv;
            }
        }

        // PV: probs(16x32) x Vt -> accumulate O (16x128)
        Frag fa;
        {
            const uint4* pp = reinterpret_cast<const uint4*>(Ps[wave] + l16 * 32);
            fa.q[0] = pp[half];
            fa.q[1] = pp[half + 2];
        }
#pragma unroll
        for (int nc = 0; nc < 8; ++nc) {
            Frag fb;
            const uint4* bp = reinterpret_cast<const uint4*>(
                Vt + (nc * 16 + l16) * 32);
            fb.q[0] = bp[half];
            fb.q[1] = bp[half + 2];
            acc[nc] = __builtin_amdgcn_wmma_f32_16x16x32_bf16(
                false, fa.v, false, fb.v, (short)0, acc[nc], false, false);
        }
        __syncthreads();
    }

    // normalize and store O (bf16 [B,S,H])
#pragma unroll
    for (int nc = 0; nc < 8; ++nc) {
#pragma unroll
        for (int r = 0; r < 8; ++r) {
            int qrow = qg + r;
            float v = acc[nc][r] / lrow[r];
            ob[(size_t)(b * Sc + qrow) * Hc + (size_t)h * HDc + nc * 16 + l16] = (__bf16)v;
        }
    }
}

// ---------------- SiLU(g) * u ----------------
__global__ void silu_mul_kernel(const __bf16* __restrict__ g,
                                const __bf16* __restrict__ u,
                                __bf16* __restrict__ o, size_t n) {
    size_t i = (size_t)blockIdx.x * blockDim.x + threadIdx.x;
    size_t stride = (size_t)gridDim.x * blockDim.x;
    for (; i < n; i += stride) {
        float gv = (float)g[i];
        float r = gv / (1.f + __expf(-gv)) * (float)u[i];
        o[i] = (__bf16)r;
    }
}

// ---------------- host launch ----------------
extern "C" void kernel_launch(void* const* d_in, const int* in_sizes, int n_in,
                              void* d_out, int out_size, void* d_ws, size_t ws_size,
                              hipStream_t stream) {
    const float* hs   = (const float*)d_in[0];
    const float* hsp  = (const float*)d_in[1];
    const float* mm   = (const float*)d_in[2];
    const float* mp   = (const float*)d_in[3];
    const int*   pid  = (const int*)d_in[4];
    const int*   pidp = (const int*)d_in[5];
    const float* Wq   = (const float*)d_in[6];
    const float* Wk   = (const float*)d_in[7];
    const float* Wv   = (const float*)d_in[8];
    const float* Wo   = (const float*)d_in[9];
    const float* Wg   = (const float*)d_in[10];
    const float* Wu   = (const float*)d_in[11];
    const float* Wd   = (const float*)d_in[12];
    const float* ln1  = (const float*)d_in[13];
    const float* ln2  = (const float*)d_in[14];
    float* outp = (float*)d_out;

    const size_t NT = (size_t)Bc * Sc;        // 4352 combined tokens
    const size_t HH = (size_t)Hc * Hc;
    const size_t IH = (size_t)Ic * Hc;

    char* ws = (char*)d_ws;
    size_t off = 0;
    auto take = [&](size_t bytes) -> void* {
        void* p = ws + off;
        off = (off + bytes + 255) & ~(size_t)255;
        return p;
    };

    __bf16* xn   = (__bf16*)take(NT * Hc * 2);
    __bf16* wqb  = (__bf16*)take(HH * 2);
    __bf16* wkb  = (__bf16*)take(HH * 2);
    __bf16* wvb  = (__bf16*)take(HH * 2);
    __bf16* wob  = (__bf16*)take(HH * 2);
    __bf16* wgb  = (__bf16*)take(IH * 2);
    __bf16* wub  = (__bf16*)take(IH * 2);
    __bf16* wdb  = (__bf16*)take(IH * 2);
    __bf16* qbuf = (__bf16*)take(NT * Hc * 2);
    __bf16* kbuf = (__bf16*)take(NT * Hc * 2);
    __bf16* vbuf = (__bf16*)take(NT * Hc * 2);
    __bf16* aob  = (__bf16*)take(NT * Hc * 2);
    float*  hs1  = (float*)take(NT * Hc * 4);
    __bf16* x2n  = (__bf16*)take(NT * Hc * 2);
    __bf16* gbuf = (__bf16*)take(NT * Ic * 2);
    __bf16* ubuf = (__bf16*)take(NT * Ic * 2);
    __bf16* hbuf = (__bf16*)take(NT * Ic * 2);

    // 1) cast weights to bf16
    cast_f32_bf16_kernel<<<4096, 256, 0, stream>>>(Wq, wqb, HH);
    cast_f32_bf16_kernel<<<4096, 256, 0, stream>>>(Wk, wkb, HH);
    cast_f32_bf16_kernel<<<4096, 256, 0, stream>>>(Wv, wvb, HH);
    cast_f32_bf16_kernel<<<4096, 256, 0, stream>>>(Wo, wob, HH);
    cast_f32_bf16_kernel<<<4096, 256, 0, stream>>>(Wg, wgb, IH);
    cast_f32_bf16_kernel<<<4096, 256, 0, stream>>>(Wu, wub, IH);
    cast_f32_bf16_kernel<<<4096, 256, 0, stream>>>(Wd, wdb, IH);

    // 2) RMSNorm 1 (split inputs) -> bf16
    rmsnorm_cast_kernel<<<(int)NT, 256, 0, stream>>>(hs, hsp, ln1, xn, 1);

    // 3) QKV projections (M=4352, N=2048, K=2048)
    dim3 gHH(Hc / 128, (int)(NT / 128));
    gemm_bf16_wmma_kernel<0><<<gHH, 256, 0, stream>>>(
        xn, wqb, (int)NT, Hc, Hc, qbuf, nullptr, nullptr, nullptr, nullptr, nullptr);
    gemm_bf16_wmma_kernel<0><<<gHH, 256, 0, stream>>>(
        xn, wkb, (int)NT, Hc, Hc, kbuf, nullptr, nullptr, nullptr, nullptr, nullptr);
    gemm_bf16_wmma_kernel<0><<<gHH, 256, 0, stream>>>(
        xn, wvb, (int)NT, Hc, Hc, vbuf, nullptr, nullptr, nullptr, nullptr, nullptr);

    // 4) RoPE on q,k
    size_t nrope = NT * NHc * (HDc / 2);
    int rblk = (int)((nrope + 255) / 256);
    rope_kernel<<<rblk, 256, 0, stream>>>(qbuf, pid, pidp);
    rope_kernel<<<rblk, 256, 0, stream>>>(kbuf, pid, pidp);

    // 5) flash attention
    dim3 gA(Sc / 64, NHc, Bc);
    attn_wmma_kernel<<<gA, 128, 0, stream>>>(qbuf, kbuf, vbuf, mm, mp, aob);

    // 6) O projection + residual -> hs1 (f32)
    gemm_bf16_wmma_kernel<1><<<gHH, 256, 0, stream>>>(
        aob, wob, (int)NT, Hc, Hc, nullptr, hs1, hs, hsp, nullptr, nullptr);

    // 7) RMSNorm 2 -> bf16
    rmsnorm_cast_kernel<<<(int)NT, 256, 0, stream>>>(hs1, nullptr, ln2, x2n, 0);

    // 8) gate / up projections (N=5632)
    dim3 gHI(Ic / 128, (int)(NT / 128));
    gemm_bf16_wmma_kernel<0><<<gHI, 256, 0, stream>>>(
        x2n, wgb, (int)NT, Ic, Hc, gbuf, nullptr, nullptr, nullptr, nullptr, nullptr);
    gemm_bf16_wmma_kernel<0><<<gHI, 256, 0, stream>>>(
        x2n, wub, (int)NT, Ic, Hc, ubuf, nullptr, nullptr, nullptr, nullptr, nullptr);

    // 9) SiLU(g)*u
    silu_mul_kernel<<<8192, 256, 0, stream>>>(gbuf, ubuf, hbuf, NT * Ic);

    // 10) down projection + residual -> split final output
    gemm_bf16_wmma_kernel<2><<<gHH, 256, 0, stream>>>(
        hbuf, wdb, (int)NT, Hc, Ic, nullptr, nullptr, nullptr, nullptr, hs1, outp);

    (void)in_sizes; (void)n_in; (void)out_size; (void)ws_size;
}
